// Tuner_9311489098238
// MI455X (gfx1250) — matrix operand order repaired and checked by reading, not compiled, not run on hardware
//
#include <hip/hip_runtime.h>

// ---------------------------------------------------------------------------
// Sizes
// ---------------------------------------------------------------------------
#define VOCAB   4101
#define KP1     4160          // 4101 padded up to multiple of 64 (K-step)
#define LSEQ    4096
#define NROWS   3072          // 1024 * 3
#define H1      768
#define H2      256
#define OUT_SEC (1024 * 256)  // elements per output section
#define KSTEP   64
#define LDST    72            // padded LDS row stride (bf16 elems) for 64-wide K tile

typedef __attribute__((ext_vector_type(16))) __bf16 bf16x16;
typedef __attribute__((ext_vector_type(8)))  float  f32x8;

// ---------------------------------------------------------------------------
// bf16 helpers (round-to-nearest-even), stored as unsigned short
// ---------------------------------------------------------------------------
static __device__ __forceinline__ unsigned short f2bf(float f) {
    unsigned u = __float_as_uint(f);
    unsigned r = u + 0x7FFFu + ((u >> 16) & 1u);
    return (unsigned short)(r >> 16);
}
static __device__ __forceinline__ float bf2f(unsigned short h) {
    return __uint_as_float(((unsigned)h) << 16);
}

// ---------------------------------------------------------------------------
// CDNA5 async global->LDS copy (ASYNCcnt path) + wait
// LDS operand VGPR = wave-relative LDS byte offset = low 32 bits of the flat
// shared-memory pointer (flat LDS aperture keeps offset in addr[31:0]).
// ---------------------------------------------------------------------------
static __device__ __forceinline__ unsigned lds_off(const void* p) {
    return (unsigned)(unsigned long long)p;
}
static __device__ __forceinline__ void async_b128(unsigned ldsOff, const void* g) {
    asm volatile("global_load_async_to_lds_b128 %0, %1, off"
                 :: "v"(ldsOff), "v"(g) : "memory");
}
template <int N>
static __device__ __forceinline__ void wait_async() {
    asm volatile("s_wait_asynccnt %0" :: "i"(N) : "memory");
}

// ---------------------------------------------------------------------------
// WMMA wrapper: D = A(16x32 bf16) x B(32x16 bf16) + C(f32)
// ---------------------------------------------------------------------------
static __device__ __forceinline__ f32x8 wmma_bf16(bf16x16 a, bf16x16 b, f32x8 c) {
    return __builtin_amdgcn_wmma_f32_16x16x32_bf16(
        false, a, false, b, (short)0, c, false, false);
}

// A fragment: lane L -> row M = L%16; lanes 0-15 hold K[0..7],[16..23],
// lanes 16-31 hold K[8..15],[24..31]   (CDNA5 16-bit A 16x32 layout)
static __device__ __forceinline__ bf16x16
load_fragA(const unsigned short* lds, int lane) {
    int m = lane & 15, g = lane >> 4;
    const unsigned short* p = lds + m * LDST + g * 8;
    bf16x16 a;
    ((uint4*)&a)[0] = *(const uint4*)(p);        // K = 8g .. 8g+7
    ((uint4*)&a)[1] = *(const uint4*)(p + 16);   // K = 16+8g .. 16+8g+7
    return a;
}

// B fragment: lane L -> col N = L%16; lanes 0-15 hold K[0..7],[16..23],
// lanes 16-31 hold K[8..15],[24..31]; LDS tile stored [n][k] (k contiguous)
static __device__ __forceinline__ bf16x16
load_fragB(const unsigned short* lds, int lane) {
    int n = lane & 15, g = lane >> 4;
    const unsigned short* p = lds + n * LDST + g * 8;
    bf16x16 b;
    ((uint4*)&b)[0] = *(const uint4*)(p);
    ((uint4*)&b)[1] = *(const uint4*)(p + 16);
    return b;
}

// ---------------------------------------------------------------------------
// Prep: W1 [4101][768] f32 -> transposed, K-padded bf16 hi/lo [768][4160]
// ---------------------------------------------------------------------------
__global__ __launch_bounds__(256) void prep_w1_kernel(
    const float* __restrict__ W1, unsigned short* __restrict__ hi,
    unsigned short* __restrict__ lo) {
    int idx = blockIdx.x * 256 + threadIdx.x;
    if (idx >= H1 * KP1) return;
    int n = idx / KP1, k = idx % KP1;
    float w = (k < VOCAB) ? W1[k * H1 + n] : 0.0f;
    unsigned short h = f2bf(w);
    hi[idx] = h;
    lo[idx] = f2bf(w - bf2f(h));
}

// Prep: W2 [768][256] f32 -> transposed bf16 hi/lo [256][768]
__global__ __launch_bounds__(256) void prep_w2_kernel(
    const float* __restrict__ W2, unsigned short* __restrict__ hi,
    unsigned short* __restrict__ lo) {
    int idx = blockIdx.x * 256 + threadIdx.x;   // H2*H1 elements
    int n = idx / H1, k = idx % H1;
    float w = W2[k * H2 + n];
    unsigned short h = f2bf(w);
    hi[idx] = h;
    lo[idx] = f2bf(w - bf2f(h));
}

// ---------------------------------------------------------------------------
// Histogram: one workgroup per row; LDS integer atomics; emits exact integer
// counts as bf16 (counts are small ints -> exactly representable) + row scale
// ---------------------------------------------------------------------------
__global__ __launch_bounds__(256) void hist_kernel(
    const int* __restrict__ X, unsigned short* __restrict__ counts_bf,
    float* __restrict__ row_scale) {
    __shared__ unsigned cnt[KP1];
    const int row = blockIdx.x;
    const int t   = threadIdx.x;

    for (int k = t; k < KP1; k += 256) cnt[k] = 0u;
    __syncthreads();

    const int* xr = X + (size_t)row * LSEQ;
    for (int i = t; i < LSEQ; i += 256) {
        unsigned tok = (unsigned)xr[i];      // tok in [0, VOCAB)
        atomicAdd(&cnt[tok], 1u);
    }
    __syncthreads();

    if (t == 0) {
        unsigned ign = cnt[0] + cnt[1] + cnt[2] + cnt[3] + cnt[4];
        unsigned kept = (unsigned)LSEQ - ign;
        row_scale[row] = kept ? (10.0f / (float)kept) : 0.0f;
    }
    unsigned short* dst = counts_bf + (size_t)row * KP1;
    for (int k = t; k < KP1; k += 256) {
        unsigned c = (k < 5 || k >= VOCAB) ? 0u : cnt[k];
        dst[k] = f2bf((float)c);
    }
}

// ---------------------------------------------------------------------------
// GEMM1: h = relu( (counts @ (W1hi+W1lo)) * scale_row + b1 )
//        stored as bf16 hi/lo pair [3072][768]
// 256 thr = 8 waves (4x2); 64x64 output tile; K-step 64; double-buffered
// async global->LDS copies overlapped with WMMA.
// ---------------------------------------------------------------------------
__global__ __launch_bounds__(256) void gemm1_kernel(
    const unsigned short* __restrict__ A,     // [3072][4160] bf16 counts
    const unsigned short* __restrict__ Bhi,   // [768][4160]
    const unsigned short* __restrict__ Blo,   // [768][4160]
    const float* __restrict__ row_scale,      // [3072]
    const float* __restrict__ b1,             // [768]
    unsigned short* __restrict__ Hhi,         // [3072][768]
    unsigned short* __restrict__ Hlo) {
    __shared__ unsigned short sA [2][64 * LDST];
    __shared__ unsigned short sBh[2][64 * LDST];
    __shared__ unsigned short sBl[2][64 * LDST];

    const int t = threadIdx.x, lane = t & 31, wave = t >> 5;
    const int wm = wave >> 1, wn = wave & 1;
    const int mBase = blockIdx.y * 64, nBase = blockIdx.x * 64;
    const int lr = t >> 2;               // coop-load row 0..63
    const int e0 = (t & 3) * 8;          // elem offsets: 2x16B chunks per row
    const int e1 = e0 + 32;

    f32x8 acc0 = {}, acc1 = {};

    auto issue = [&](int buf, int k0) {
        size_t ga = (size_t)(mBase + lr) * KP1 + k0;
        size_t gb = (size_t)(nBase + lr) * KP1 + k0;
        async_b128(lds_off(&sA [buf][lr * LDST + e0]), &A  [ga + e0]);
        async_b128(lds_off(&sA [buf][lr * LDST + e1]), &A  [ga + e1]);
        async_b128(lds_off(&sBh[buf][lr * LDST + e0]), &Bhi[gb + e0]);
        async_b128(lds_off(&sBh[buf][lr * LDST + e1]), &Bhi[gb + e1]);
        async_b128(lds_off(&sBl[buf][lr * LDST + e0]), &Blo[gb + e0]);
        async_b128(lds_off(&sBl[buf][lr * LDST + e1]), &Blo[gb + e1]);
    };

    issue(0, 0);
    int cur = 0;
    for (int it = 0; it < KP1 / KSTEP; ++it) {     // 65 iterations
        if (it + 1 < KP1 / KSTEP) {
            issue(cur ^ 1, (it + 1) * KSTEP);      // prefetch next tile
            wait_async<6>();                       // current tile's 6 done
        } else {
            wait_async<0>();
        }
        __syncthreads();                           // all waves: tile ready
#pragma unroll
        for (int kk = 0; kk < KSTEP; kk += 32) {
            bf16x16 a   = load_fragA(&sA [cur][wm * 16 * LDST + kk], lane);
            bf16x16 bh0 = load_fragB(&sBh[cur][(wn * 32 +  0) * LDST + kk], lane);
            bf16x16 bl0 = load_fragB(&sBl[cur][(wn * 32 +  0) * LDST + kk], lane);
            bf16x16 bh1 = load_fragB(&sBh[cur][(wn * 32 + 16) * LDST + kk], lane);
            bf16x16 bl1 = load_fragB(&sBl[cur][(wn * 32 + 16) * LDST + kk], lane);
            acc0 = wmma_bf16(a, bh0, acc0);
            acc0 = wmma_bf16(a, bl0, acc0);   // hi+lo weights -> ~fp32 accuracy
            acc1 = wmma_bf16(a, bh1, acc1);
            acc1 = wmma_bf16(a, bl1, acc1);
        }
        __syncthreads();                           // readers done -> refillable
        cur ^= 1;
    }

    const int col0    = nBase + wn * 32 + (lane & 15);
    const int rowBase = mBase + wm * 16 + ((lane >> 4) << 3);
    const float bias0 = b1[col0], bias1 = b1[col0 + 16];
#pragma unroll
    for (int r = 0; r < 8; ++r) {
        int row = rowBase + r;
        float s  = row_scale[row];
        float x0 = fmaxf(acc0[r] * s + bias0, 0.0f);
        float x1 = fmaxf(acc1[r] * s + bias1, 0.0f);
        unsigned short h0 = f2bf(x0), h1 = f2bf(x1);
        size_t o = (size_t)row * H1;
        Hhi[o + col0]      = h0;
        Hlo[o + col0]      = f2bf(x0 - bf2f(h0));
        Hhi[o + col0 + 16] = h1;
        Hlo[o + col0 + 16] = f2bf(x1 - bf2f(h1));
    }
}

// ---------------------------------------------------------------------------
// GEMM2: out = h @ (W2hi+W2lo) + b2, scattered to the 3 output sections.
// (Ahi+Alo)(Bhi+Blo) ~= Ahi*Bhi + Ahi*Blo + Alo*Bhi  (lo*lo dropped)
// ---------------------------------------------------------------------------
__global__ __launch_bounds__(256) void gemm2_kernel(
    const unsigned short* __restrict__ Ahi,   // [3072][768]
    const unsigned short* __restrict__ Alo,
    const unsigned short* __restrict__ Bhi,   // [256][768]
    const unsigned short* __restrict__ Blo,
    const float* __restrict__ b2,             // [256]
    float* __restrict__ out) {                // 3 x [1024][256]
    __shared__ unsigned short sAh[2][64 * LDST];
    __shared__ unsigned short sAl[2][64 * LDST];
    __shared__ unsigned short sBh[2][64 * LDST];
    __shared__ unsigned short sBl[2][64 * LDST];

    const int t = threadIdx.x, lane = t & 31, wave = t >> 5;
    const int wm = wave >> 1, wn = wave & 1;
    const int mBase = blockIdx.y * 64, nBase = blockIdx.x * 64;
    const int lr = t >> 2;
    const int e0 = (t & 3) * 8;
    const int e1 = e0 + 32;

    f32x8 acc0 = {}, acc1 = {};

    auto issue = [&](int buf, int k0) {
        size_t ga = (size_t)(mBase + lr) * H1 + k0;
        size_t gb = (size_t)(nBase + lr) * H1 + k0;
        async_b128(lds_off(&sAh[buf][lr * LDST + e0]), &Ahi[ga + e0]);
        async_b128(lds_off(&sAh[buf][lr * LDST + e1]), &Ahi[ga + e1]);
        async_b128(lds_off(&sAl[buf][lr * LDST + e0]), &Alo[ga + e0]);
        async_b128(lds_off(&sAl[buf][lr * LDST + e1]), &Alo[ga + e1]);
        async_b128(lds_off(&sBh[buf][lr * LDST + e0]), &Bhi[gb + e0]);
        async_b128(lds_off(&sBh[buf][lr * LDST + e1]), &Bhi[gb + e1]);
        async_b128(lds_off(&sBl[buf][lr * LDST + e0]), &Blo[gb + e0]);
        async_b128(lds_off(&sBl[buf][lr * LDST + e1]), &Blo[gb + e1]);
    };

    issue(0, 0);
    int cur = 0;
    for (int it = 0; it < H1 / KSTEP; ++it) {      // 12 iterations
        if (it + 1 < H1 / KSTEP) {
            issue(cur ^ 1, (it + 1) * KSTEP);
            wait_async<8>();
        } else {
            wait_async<0>();
        }
        __syncthreads();
#pragma unroll
        for (int kk = 0; kk < KSTEP; kk += 32) {
            bf16x16 ah  = load_fragA(&sAh[cur][wm * 16 * LDST + kk], lane);
            bf16x16 al  = load_fragA(&sAl[cur][wm * 16 * LDST + kk], lane);
            bf16x16 bh0 = load_fragB(&sBh[cur][(wn * 32 +  0) * LDST + kk], lane);
            bf16x16 bl0 = load_fragB(&sBl[cur][(wn * 32 +  0) * LDST + kk], lane);
            bf16x16 bh1 = load_fragB(&sBh[cur][(wn * 32 + 16) * LDST + kk], lane);
            bf16x16 bl1 = load_fragB(&sBl[cur][(wn * 32 + 16) * LDST + kk], lane);
            acc0 = wmma_bf16(ah, bh0, acc0);
            acc0 = wmma_bf16(ah, bl0, acc0);
            acc0 = wmma_bf16(al, bh0, acc0);
            acc1 = wmma_bf16(ah, bh1, acc1);
            acc1 = wmma_bf16(ah, bl1, acc1);
            acc1 = wmma_bf16(al, bh1, acc1);
        }
        __syncthreads();
        cur ^= 1;
    }

    const int col0    = nBase + wn * 32 + (lane & 15);
    const int rowBase = mBase + wm * 16 + ((lane >> 4) << 3);
    const float bias0 = b2[col0], bias1 = b2[col0 + 16];
#pragma unroll
    for (int r = 0; r < 8; ++r) {
        int row = rowBase + r;               // flat row = batch*3 + section
        int sec = row % 3, b = row / 3;
        size_t o = (size_t)sec * OUT_SEC + (size_t)b * H2;
        out[o + col0]      = acc0[r] + bias0;
        out[o + col0 + 16] = acc1[r] + bias1;
    }
}

// ---------------------------------------------------------------------------
// Launch
// ---------------------------------------------------------------------------
extern "C" void kernel_launch(void* const* d_in, const int* in_sizes, int n_in,
                              void* d_out, int out_size, void* d_ws, size_t ws_size,
                              hipStream_t stream) {
    const int*   X  = (const int*)  d_in[0];   // [1024][3][4096]
    const float* W1 = (const float*)d_in[1];   // [4101][768]
    const float* b1 = (const float*)d_in[2];   // [768]
    const float* W2 = (const float*)d_in[3];   // [768][256]
    const float* b2 = (const float*)d_in[4];   // [256]
    float*       out = (float*)d_out;          // 3 x [1024][256]

    char*  ws = (char*)d_ws;
    size_t o  = 0;
    auto carve = [&](size_t bytes) -> void* {
        void* p = ws + o;
        o = (o + bytes + 255) & ~(size_t)255;
        return p;
    };
    unsigned short* A1    = (unsigned short*)carve((size_t)NROWS * KP1 * 2);
    float*          scale = (float*)         carve((size_t)NROWS * 4);
    unsigned short* w1hi  = (unsigned short*)carve((size_t)H1 * KP1 * 2);
    unsigned short* w1lo  = (unsigned short*)carve((size_t)H1 * KP1 * 2);
    unsigned short* w2hi  = (unsigned short*)carve((size_t)H2 * H1 * 2);
    unsigned short* w2lo  = (unsigned short*)carve((size_t)H2 * H1 * 2);
    unsigned short* hhi   = (unsigned short*)carve((size_t)NROWS * H1 * 2);
    unsigned short* hlo   = (unsigned short*)carve((size_t)NROWS * H1 * 2);
    (void)ws_size; (void)in_sizes; (void)n_in; (void)out_size;

    prep_w1_kernel<<<(H1 * KP1 + 255) / 256, 256, 0, stream>>>(W1, w1hi, w1lo);
    prep_w2_kernel<<<(H2 * H1) / 256,         256, 0, stream>>>(W2, w2hi, w2lo);
    hist_kernel   <<<NROWS,                   256, 0, stream>>>(X, A1, scale);
    gemm1_kernel  <<<dim3(H1 / 64, NROWS / 64), 256, 0, stream>>>(
        A1, w1hi, w1lo, scale, b1, hhi, hlo);
    gemm2_kernel  <<<dim3(H2 / 64, NROWS / 64), 256, 0, stream>>>(
        hhi, hlo, w2hi, w2lo, b2, out);
}
